// SegmentLinear_14508399526013
// MI455X (gfx1250) — 8-bit matrix operand order repaired and checked by reading, compiled, not run on hardware
//
#include <hip/hip_runtime.h>
#include <hip/hip_bf16.h>

typedef int   v8i __attribute__((ext_vector_type(8)));
typedef float v8f __attribute__((ext_vector_type(8)));

constexpr int N  = 4096;
constexpr int K  = 4096;
constexpr int O  = 4096;
constexpr int CHUNKS = 4;
constexpr int CS = K / CHUNKS;
constexpr float QMAX = 127.0f;
constexpr float EPS  = 1e-8f;

__global__ void seglin_init(unsigned int* __restrict__ amax) {
    if (threadIdx.x < 2 * CHUNKS) amax[threadIdx.x] = 0u;
}

__global__ __launch_bounds__(256)
void seglin_amax(const float4* __restrict__ x4, const float4* __restrict__ w4,
                 unsigned int* __restrict__ amax) {
    const int tc = blockIdx.y;
    const int c  = tc & 3;
    const float4* src = (tc < CHUNKS) ? x4 : w4;

    float m = 0.0f;
    const unsigned total = (unsigned)N * (CS / 4);
    for (unsigned i = blockIdx.x * blockDim.x + threadIdx.x; i < total;
         i += gridDim.x * blockDim.x) {
        unsigned row = i >> 8, col = i & 255u;
        float4 v = src[(size_t)row * (K / 4) + (unsigned)c * (CS / 4) + col];
        m = fmaxf(m, fmaxf(fmaxf(fabsf(v.x), fabsf(v.y)),
                           fmaxf(fabsf(v.z), fabsf(v.w))));
    }
    __shared__ float sm[256];
    sm[threadIdx.x] = m;
    __syncthreads();
    for (int s = 128; s > 0; s >>= 1) {
        if ((int)threadIdx.x < s) sm[threadIdx.x] = fmaxf(sm[threadIdx.x], sm[threadIdx.x + s]);
        __syncthreads();
    }
    if (threadIdx.x == 0) atomicMax(&amax[tc], __float_as_uint(sm[0]));
}

__device__ __forceinline__ int q8(float v, float inv) {
    int q = __float2int_rn(v * inv);
    return min(127, max(-127, q));
}

__global__ __launch_bounds__(256)
void seglin_quant(const float4* __restrict__ x4, const float4* __restrict__ w4,
                  const unsigned int* __restrict__ amax,
                  unsigned int* __restrict__ qx4, unsigned int* __restrict__ qw4) {
    const int t   = blockIdx.y;
    const unsigned idx = blockIdx.x * blockDim.x + threadIdx.x;
    const float4* src = t ? w4 : x4;
    unsigned int* dst = t ? qw4 : qx4;

    const unsigned col4 = idx & (K / 4 - 1);
    const int c = (int)(col4 >> 8);
    const float am    = __uint_as_float(amax[t * CHUNKS + c]);
    const float scale = fmaxf(am * (1.0f / QMAX), EPS);
    const float inv   = 1.0f / scale;

    float4 v = src[idx];
    unsigned p = (unsigned)(q8(v.x, inv) & 0xFF)
               | ((unsigned)(q8(v.y, inv) & 0xFF) << 8)
               | ((unsigned)(q8(v.z, inv) & 0xFF) << 16)
               | ((unsigned)(q8(v.w, inv) & 0xFF) << 24);
    dst[idx] = p;
}

__device__ __forceinline__ v8i load_a_frag(const signed char* p) {
    int2 x0 = *(const int2*)(p);
    int2 x1 = *(const int2*)(p + 16);
    int2 x2 = *(const int2*)(p + 32);
    int2 x3 = *(const int2*)(p + 48);
    v8i a; a[0]=x0.x; a[1]=x0.y; a[2]=x1.x; a[3]=x1.y;
           a[4]=x2.x; a[5]=x2.y; a[6]=x3.x; a[7]=x3.y;
    return a;
}

__device__ __forceinline__ v8i load_b_frag(const signed char* p) {
    int2 y0 = *(const int2*)(p);
    int2 y1 = *(const int2*)(p + 16);
    int2 y2 = *(const int2*)(p + 32);
    int2 y3 = *(const int2*)(p + 48);
    v8i b; b[0]=y0.x; b[1]=y0.y; b[2]=y1.x; b[3]=y1.y;
           b[4]=y2.x; b[5]=y2.y; b[6]=y3.x; b[7]=y3.y;
    return b;
}

__global__ __launch_bounds__(256)
void seglin_i8gemm(const signed char* __restrict__ qx,
                   const signed char* __restrict__ qw,
                   const unsigned int* __restrict__ amax,
                   float* __restrict__ out) {
    const int tid  = threadIdx.x;
    const int wave = tid >> 5;
    const int lane = tid & 31;
    const int lg   = lane >> 4;
    const int ml   = lane & 15;

    const int wm = wave & 3;
    const int wn = wave >> 2;
    const int m0 = blockIdx.x * 128 + wm * 32;
    const int o0 = blockIdx.y * 64  + wn * 32;

    const signed char* a0p = qx + (size_t)(m0 + ml)      * K + lg * 8;
    const signed char* a1p = qx + (size_t)(m0 + 16 + ml) * K + lg * 8;
    const signed char* b0p = qw + (size_t)(o0 + ml)      * K + lg * 8;
    const signed char* b1p = qw + (size_t)(o0 + 16 + ml) * K + lg * 8;

    v8f f00 = {}, f01 = {}, f10 = {}, f11 = {};

    for (int c = 0; c < CHUNKS; ++c) {
        const float sx = fmaxf(__uint_as_float(amax[c])          * (1.0f / QMAX), EPS);
        const float sw = fmaxf(__uint_as_float(amax[CHUNKS + c]) * (1.0f / QMAX), EPS);
        const float sf = sx * sw;

        v8i i00 = {}, i01 = {}, i10 = {}, i11 = {};
        const int kbase = c * CS;
        #pragma unroll 4
        for (int ks = 0; ks < CS / 64; ++ks) {
            const int kb = kbase + ks * 64;
            v8i a0 = load_a_frag(a0p + kb);
            v8i a1 = load_a_frag(a1p + kb);
            v8i b0 = load_b_frag(b0p + kb);
            v8i b1 = load_b_frag(b1p + kb);
            i00 = __builtin_amdgcn_wmma_i32_16x16x64_iu8(true, a0, true, b0, i00, false, false);
            i01 = __builtin_amdgcn_wmma_i32_16x16x64_iu8(true, a0, true, b1, i01, false, false);
            i10 = __builtin_amdgcn_wmma_i32_16x16x64_iu8(true, a1, true, b0, i10, false, false);
            i11 = __builtin_amdgcn_wmma_i32_16x16x64_iu8(true, a1, true, b1, i11, false, false);
        }
        #pragma unroll
        for (int r = 0; r < 8; ++r) {
            f00[r] += sf * (float)i00[r];
            f01[r] += sf * (float)i01[r];
            f10[r] += sf * (float)i10[r];
            f11[r] += sf * (float)i11[r];
        }
    }

    const int row0 = m0 + lg * 8;
    const int col  = o0 + ml;
    #pragma unroll
    for (int r = 0; r < 8; ++r) {
        out[(size_t)(row0 + r)      * O + col]      = f00[r];
        out[(size_t)(row0 + r)      * O + col + 16] = f01[r];
        out[(size_t)(row0 + 16 + r) * O + col]      = f10[r];
        out[(size_t)(row0 + 16 + r) * O + col + 16] = f11[r];
    }
}

extern "C" void kernel_launch(void* const* d_in, const int* in_sizes, int n_in,
                              void* d_out, int out_size, void* d_ws, size_t ws_size,
                              hipStream_t stream) {
    const float* x = (const float*)d_in[0];
    const float* w = (const float*)d_in[1];
    float* out     = (float*)d_out;

    unsigned int* amax = (unsigned int*)d_ws;
    signed char*  qx   = (signed char*)d_ws + 256;
    signed char*  qw   = qx + (size_t)N * K;

    seglin_init<<<1, 32, 0, stream>>>(amax);

    seglin_amax<<<dim3(256, 2 * CHUNKS), 256, 0, stream>>>(
        (const float4*)x, (const float4*)w, amax);

    seglin_quant<<<dim3((N * K / 4) / 256, 2), 256, 0, stream>>>(
        (const float4*)x, (const float4*)w, amax,
        (unsigned int*)qx, (unsigned int*)qw);

    seglin_i8gemm<<<dim3(N / 128, O / 64), 256, 0, stream>>>(qx, qw, amax, out);
}
